// MultiViewEntropyBottleneck_30150670418089
// MI455X (gfx1250) — matrix operand order repaired and checked by reading, not compiled, not run on hardware
//
#include <hip/hip_runtime.h>
#include <hip/hip_bf16.h>

typedef _Float16 half_t;
typedef __attribute__((ext_vector_type(16))) _Float16 v16h;
typedef __attribute__((ext_vector_type(8)))  float    v8f;

#define WMMA_F16(a, b, c) \
  __builtin_amdgcn_wmma_f32_16x16x32_f16(false, (a), false, (b), (short)0, (c), false, false)

#define NB 1024   // batch
#define NS 64     // seq
#define ND 256    // latent D

// ---------------------------------------------------------------------------
// Weight transpose + f32->f16 convert:  src (K,N) row-major f32 -> dst (N,K) f16
// ---------------------------------------------------------------------------
__global__ __launch_bounds__(256) void transpose_f16_kernel(
    const float* __restrict__ src, half_t* __restrict__ dst, int K, int N) {
  int idx = blockIdx.x * 256 + threadIdx.x;
  if (idx < K * N) {
    int n = idx % N, k = idx / N;
    dst[(size_t)n * K + k] = (half_t)src[idx];
  }
}

// ---------------------------------------------------------------------------
// seq_len[b] = sum(mask[b,:]) - 1
// ---------------------------------------------------------------------------
__global__ __launch_bounds__(256) void seqlen_kernel(
    const int* __restrict__ mask, int* __restrict__ seq_len) {
  int b = blockIdx.x * 256 + threadIdx.x;
  if (b < NB) {
    int s = 0;
    for (int i = 0; i < NS; ++i) s += mask[b * NS + i];
    seq_len[b] = s - 1;
  }
}

// ---------------------------------------------------------------------------
// Fused MLP:  X(ntok,IN) -> Linear(W1,b1) -> LN(g,be) -> GELU -> Linear(W2,b2)
// REPARAM epilogue: split mu/lv, z = l2norm(mu + eps*exp(lv/2)), KL at seq_len.
// Block = 256 threads (8 waves), 32 tokens per block (2 row tiles -> each
// B fragment feeds two WMMAs).  Dynamic LDS: 32*HID_DIM*2 bytes.
// ---------------------------------------------------------------------------
template <int IN_DIM, int HID_DIM, int OUT_DIM, bool REPARAM>
__global__ __launch_bounds__(256) void fused_mlp_kernel(
    const float* __restrict__ X,
    const half_t* __restrict__ W1t,   // (HID,IN) f16
    const float* __restrict__ b1,
    const float* __restrict__ gam,
    const float* __restrict__ bet,
    const half_t* __restrict__ W2t,   // (OUT,HID) f16
    const float* __restrict__ b2,
    const float* __restrict__ epsn,   // (ntok,ND) or null
    const int* __restrict__ seq_len,  // (NB) or null
    float* __restrict__ z_out,        // (ntok,ND) or null
    float* __restrict__ zsel,         // (NB,ND) or null
    float* __restrict__ kl_out,       // (NB) or null
    float* __restrict__ plain_out)    // (ntok,OUT) or null
{
  constexpr int NT1 = HID_DIM / 128;  // col tiles per wave, GEMM1
  constexpr int NT2 = OUT_DIM / 128;  // col tiles per wave, GEMM2

  // sH (32 x HID f16) aliases sC2 (32 x OUT f32): HID == 2*OUT
  extern __shared__ __align__(16) char smem_raw[];
  half_t* sH  = (half_t*)smem_raw;
  float*  sC2 = (float*)smem_raw;
  __shared__ float sRedA[8][32];
  __shared__ float sRedB[8][32];
  __shared__ float sMean[32];
  __shared__ float sRstd[32];

  const int tid  = threadIdx.x;
  const int wave = tid >> 5;
  const int lane = tid & 31;
  const int hl   = lane >> 4;   // half-wave select
  const int ll   = lane & 15;
  const int row0 = blockIdx.x * 32;

  // ---------------- GEMM1: H = X @ W1 ----------------
  v8f acc[2][NT1];
#pragma unroll
  for (int rt = 0; rt < 2; ++rt)
#pragma unroll
    for (int t = 0; t < NT1; ++t) acc[rt][t] = v8f{0.f,0.f,0.f,0.f,0.f,0.f,0.f,0.f};
  const int ncol0 = wave * NT1 * 16;
  for (int k0 = 0; k0 < IN_DIM; k0 += 32) {
    // A fragments: lanes 0-15 hold K{0-7,16-23}, lanes 16-31 hold K{8-15,24-31}
    v16h a[2];
#pragma unroll
    for (int rt = 0; rt < 2; ++rt) {
      const float* ap = X + (size_t)(row0 + rt * 16 + ll) * IN_DIM + k0 + (hl ? 8 : 0);
#pragma unroll
      for (int i = 0; i < 8; ++i) a[rt][i] = (half_t)ap[i];
#pragma unroll
      for (int i = 0; i < 8; ++i) a[rt][8 + i] = (half_t)ap[16 + i];
    }
#pragma unroll
    for (int t = 0; t < NT1; ++t) {
      // B fragment: lane = N column, lanes 0-15 hold K{0-7,16-23}, lanes 16-31 hold K{8-15,24-31}
      const half_t* bp = W1t + (size_t)(ncol0 + t * 16 + ll) * IN_DIM + k0 + (hl ? 8 : 0);
      v16h b;
#pragma unroll
      for (int i = 0; i < 16; ++i) b[i] = bp[i < 8 ? i : i + 8];
      acc[0][t] = WMMA_F16(a[0], b, acc[0][t]);
      acc[1][t] = WMMA_F16(a[1], b, acc[1][t]);
    }
  }

  // bias + LN statistics from registers (row = rt*16 + hl*8 + i)
  float ps[2][8], ps2[2][8];
#pragma unroll
  for (int rt = 0; rt < 2; ++rt)
#pragma unroll
    for (int i = 0; i < 8; ++i) { ps[rt][i] = 0.f; ps2[rt][i] = 0.f; }
#pragma unroll
  for (int t = 0; t < NT1; ++t) {
    float bb = b1[ncol0 + t * 16 + ll];
#pragma unroll
    for (int rt = 0; rt < 2; ++rt)
#pragma unroll
      for (int i = 0; i < 8; ++i) {
        float v = acc[rt][t][i] + bb;
        acc[rt][t][i] = v;
        ps[rt][i]  += v;
        ps2[rt][i] += v * v;
      }
  }
#pragma unroll
  for (int rt = 0; rt < 2; ++rt)
#pragma unroll
    for (int i = 0; i < 8; ++i) {
#pragma unroll
      for (int m = 1; m < 16; m <<= 1) {
        ps[rt][i]  += __shfl_xor(ps[rt][i],  m, 32);
        ps2[rt][i] += __shfl_xor(ps2[rt][i], m, 32);
      }
    }
  if (ll == 0) {
#pragma unroll
    for (int rt = 0; rt < 2; ++rt)
#pragma unroll
      for (int i = 0; i < 8; ++i) {
        sRedA[wave][rt * 16 + hl * 8 + i] = ps[rt][i];
        sRedB[wave][rt * 16 + hl * 8 + i] = ps2[rt][i];
      }
  }
  __syncthreads();
  if (tid < 32) {
    float s = 0.f, s2 = 0.f;
#pragma unroll
    for (int w = 0; w < 8; ++w) { s += sRedA[w][tid]; s2 += sRedB[w][tid]; }
    float mean = s / (float)HID_DIM;
    float var  = s2 / (float)HID_DIM - mean * mean;
    sMean[tid] = mean;
    sRstd[tid] = rsqrtf(var + 1e-5f);
  }
  __syncthreads();

  // LN + exact GELU -> sH (f16)
#pragma unroll
  for (int t = 0; t < NT1; ++t) {
    int n = ncol0 + t * 16 + ll;
    float gv = gam[n], bv = bet[n];
#pragma unroll
    for (int rt = 0; rt < 2; ++rt)
#pragma unroll
      for (int i = 0; i < 8; ++i) {
        int r = rt * 16 + hl * 8 + i;
        float x  = (acc[rt][t][i] - sMean[r]) * sRstd[r] * gv + bv;
        float ge = 0.5f * x * (1.0f + erff(x * 0.70710678118654752f));
        sH[r * HID_DIM + n] = (half_t)ge;
      }
  }
  __syncthreads();

  // ---------------- GEMM2: C2 = H @ W2 ----------------
  v8f acc2[2][NT2];
#pragma unroll
  for (int rt = 0; rt < 2; ++rt)
#pragma unroll
    for (int t = 0; t < NT2; ++t) acc2[rt][t] = v8f{0.f,0.f,0.f,0.f,0.f,0.f,0.f,0.f};
  const int mcol0 = wave * NT2 * 16;
  for (int k0 = 0; k0 < HID_DIM; k0 += 32) {
    v16h a[2];
#pragma unroll
    for (int rt = 0; rt < 2; ++rt) {
      const half_t* ap = sH + (rt * 16 + ll) * HID_DIM + k0 + (hl ? 8 : 0);
#pragma unroll
      for (int i = 0; i < 8; ++i) a[rt][i] = ap[i];
#pragma unroll
      for (int i = 0; i < 8; ++i) a[rt][8 + i] = ap[16 + i];
    }
#pragma unroll
    for (int t = 0; t < NT2; ++t) {
      const half_t* bp = W2t + (size_t)(mcol0 + t * 16 + ll) * HID_DIM + k0 + (hl ? 8 : 0);
      v16h b;
#pragma unroll
      for (int i = 0; i < 16; ++i) b[i] = bp[i < 8 ? i : i + 8];
      acc2[0][t] = WMMA_F16(a[0], b, acc2[0][t]);
      acc2[1][t] = WMMA_F16(a[1], b, acc2[1][t]);
    }
  }
  __syncthreads();   // all sH reads done; safe to overwrite with sC2
#pragma unroll
  for (int t = 0; t < NT2; ++t) {
    int n = mcol0 + t * 16 + ll;
    float bb = b2[n];
#pragma unroll
    for (int rt = 0; rt < 2; ++rt)
#pragma unroll
      for (int i = 0; i < 8; ++i)
        sC2[(rt * 16 + hl * 8 + i) * OUT_DIM + n] = acc2[rt][t][i] + bb;
  }
  __syncthreads();

  // ---------------- epilogue ----------------
  if constexpr (REPARAM) {
    constexpr int D  = OUT_DIM / 2;   // 256
    constexpr int PJ = D / 8;         // 32 elems per thread (8 threads per row)
    const int r     = tid >> 3;       // token row 0..31
    const int cg    = tid & 7;
    const int token = row0 + r;
    const int b     = token / NS;
    const int s     = token % NS;
    float zv[PJ];
    float ss = 0.f, klp = 0.f;
#pragma unroll
    for (int j = 0; j < PJ; ++j) {
      int d = cg * PJ + j;
      float mu = sC2[r * OUT_DIM + d];
      float lv = sC2[r * OUT_DIM + D + d];
      float ev = epsn[(size_t)token * D + d];
      float z  = mu + ev * expf(0.5f * lv);
      zv[j] = z;
      ss  += z * z;
      klp += 1.0f + lv - mu * mu - expf(lv);
    }
#pragma unroll
    for (int m = 1; m < 8; m <<= 1) {   // 8 threads of a row share an 8-lane group
      ss  += __shfl_xor(ss,  m, 32);
      klp += __shfl_xor(klp, m, 32);
    }
    float inv = 1.0f / fmaxf(sqrtf(ss), 1e-12f);
    bool sel = (seq_len[b] == s);
#pragma unroll
    for (int j = 0; j < PJ; ++j) {
      int d = cg * PJ + j;
      float zn = zv[j] * inv;
      z_out[(size_t)token * D + d] = zn;
      if (sel) zsel[(size_t)b * D + d] = zn;
    }
    if (sel && cg == 0) kl_out[b] = -0.5f * klp;
  } else {
    for (int idx = tid; idx < 32 * OUT_DIM; idx += 256) {
      int r = idx / OUT_DIM, n = idx % OUT_DIM;
      plain_out[(size_t)(row0 + r) * OUT_DIM + n] = sC2[idx];
    }
  }
}

// ---------------------------------------------------------------------------
// Row norms: xx[i]=sum x_i^2, yy[j]=sum y_j^2, ysy[j]=sum y_j*sy_j   (rows of 256)
// ---------------------------------------------------------------------------
__global__ __launch_bounds__(256) void rownorm_kernel(
    const float* __restrict__ x, const float* __restrict__ y,
    const float* __restrict__ sy,
    float* __restrict__ xx, float* __restrict__ yy, float* __restrict__ ysy) {
  int b = blockIdx.x, tid = threadIdx.x, wave = tid >> 5, lane = tid & 31;
  float vx = x[(size_t)b * ND + tid];
  float vy = y[(size_t)b * ND + tid];
  float vs = sy[(size_t)b * ND + tid];
  float a = vx * vx, c = vy * vy, d = vy * vs;
  __shared__ float s1[8], s2[8], s3[8];
#pragma unroll
  for (int m = 1; m < 32; m <<= 1) {
    a += __shfl_xor(a, m, 32);
    c += __shfl_xor(c, m, 32);
    d += __shfl_xor(d, m, 32);
  }
  if (lane == 0) { s1[wave] = a; s2[wave] = c; s3[wave] = d; }
  __syncthreads();
  if (tid == 0) {
    float ta = 0, tc = 0, td = 0;
#pragma unroll
    for (int w = 0; w < 8; ++w) { ta += s1[w]; tc += s2[w]; td += s3[w]; }
    xx[b] = ta; yy[b] = tc; ysy[b] = td;
  }
}

// ---------------------------------------------------------------------------
// Pairwise: one 16x16 output tile per wave; 3 WMMA GEMMs over K=256.
// dist = max(xx+yy-2*x.y, 0); also store sx.sy and x.sy grams.
// ---------------------------------------------------------------------------
__global__ __launch_bounds__(256) void pairwise_kernel(
    const float* __restrict__ x, const float* __restrict__ sx,
    const float* __restrict__ y, const float* __restrict__ sy,
    const float* __restrict__ xx, const float* __restrict__ yy,
    float* __restrict__ dist, float* __restrict__ gss, float* __restrict__ gxsy) {
  const int tid = threadIdx.x, wave = tid >> 5, lane = tid & 31;
  const int hl = lane >> 4, ll = lane & 15;
  const int tile = blockIdx.x * 8 + wave;       // 4096 tiles
  const int ti = tile >> 6, tj = tile & 63;
  v8f a_xy  = v8f{0.f,0.f,0.f,0.f,0.f,0.f,0.f,0.f};
  v8f a_ss  = a_xy, a_xsy = a_xy;
  for (int k0 = 0; k0 < ND; k0 += 32) {
    const float* px  = x  + (size_t)(ti * 16 + ll) * ND + k0 + (hl ? 8 : 0);
    const float* psx = sx + (size_t)(ti * 16 + ll) * ND + k0 + (hl ? 8 : 0);
    v16h ax, asx;
#pragma unroll
    for (int i = 0; i < 8; ++i) { ax[i] = (half_t)px[i];  asx[i] = (half_t)psx[i]; }
#pragma unroll
    for (int i = 0; i < 8; ++i) { ax[8+i] = (half_t)px[16+i]; asx[8+i] = (half_t)psx[16+i]; }
    const float* py  = y  + (size_t)(tj * 16 + ll) * ND + k0 + (hl ? 8 : 0);
    const float* psy = sy + (size_t)(tj * 16 + ll) * ND + k0 + (hl ? 8 : 0);
    v16h by, bsy;
#pragma unroll
    for (int i = 0; i < 16; ++i) { by[i] = (half_t)py[i < 8 ? i : i + 8]; bsy[i] = (half_t)psy[i < 8 ? i : i + 8]; }
    a_xy  = WMMA_F16(ax,  by,  a_xy);
    a_ss  = WMMA_F16(asx, bsy, a_ss);
    a_xsy = WMMA_F16(ax,  bsy, a_xsy);
  }
#pragma unroll
  for (int i = 0; i < 8; ++i) {
    int I = ti * 16 + hl * 8 + i;
    int J = tj * 16 + ll;
    size_t o = (size_t)I * NB + J;
    float d = fmaxf(xx[I] + yy[J] - 2.0f * a_xy[i], 0.0f);
    dist[o] = d;
    gss[o]  = a_ss[i];
    gxsy[o] = a_xsy[i];
  }
}

// ---------------------------------------------------------------------------
// Exact median of dist[:512,:512] via 2x radix-select on float bits (all >= 0),
// then bandwidth -> h2.  Single block of 1024 threads.
// ---------------------------------------------------------------------------
__global__ __launch_bounds__(1024) void bandwidth_kernel(
    const float* __restrict__ dist, float* __restrict__ h2out) {
  const int tid = threadIdx.x;
  __shared__ unsigned swv[32];
  __shared__ unsigned sc0;
  float vv[2];
  for (int sel = 0; sel < 2; ++sel) {
    unsigned k = 131071u + (unsigned)sel;   // two middle order statistics of 262144
    unsigned prefix = 0u;
    for (int bit = 30; bit >= 0; --bit) {
      unsigned himask = 0xFFFFFFFFu << (bit + 1);
      unsigned cnt = 0;
      for (int idx = tid; idx < 512 * 512; idx += 1024) {
        unsigned u = __float_as_uint(dist[(size_t)(idx >> 9) * NB + (idx & 511)]);
        if (((u & himask) == prefix) && (((u >> bit) & 1u) == 0u)) cnt++;
      }
#pragma unroll
      for (int m = 1; m < 32; m <<= 1) cnt += __shfl_xor(cnt, m, 32);
      if ((tid & 31) == 0) swv[tid >> 5] = cnt;
      __syncthreads();
      if (tid == 0) {
        unsigned tot = 0;
        for (int w = 0; w < 32; ++w) tot += swv[w];
        sc0 = tot;
      }
      __syncthreads();
      unsigned c0 = sc0;
      if (k >= c0) { k -= c0; prefix |= (1u << bit); }
      __syncthreads();
    }
    vv[sel] = __uint_as_float(prefix);
  }
  if (tid == 0) {
    float med = 0.5f * (vv[0] + vv[1]);
    float bw  = sqrtf(med * 0.5f);              // * BW_FACTOR(=1)
    bw = fminf(fmaxf(bw, 0.1f), 10.0f);
    h2out[0] = bw * bw;
  }
}

// ---------------------------------------------------------------------------
// Stein-kernel terms; per-block partial sums (deterministic, no atomics)
// ---------------------------------------------------------------------------
__global__ __launch_bounds__(256) void align_kernel(
    const float* __restrict__ dist, const float* __restrict__ gss,
    const float* __restrict__ gxsy, const float* __restrict__ ysy,
    const float* __restrict__ h2p, float* __restrict__ partial) {
  const float h2    = h2p[0];
  const float ih2   = 1.0f / h2;
  const float i2h2  = 0.5f * ih2;
  const float ih4   = ih2 * ih2;
  const float dimh  = (float)ND * ih2;
  size_t base = (size_t)blockIdx.x * 1024;
  float s = 0.f;
#pragma unroll
  for (int t = 0; t < 4; ++t) {
    size_t idx = base + threadIdx.x + t * 256;
    int j = (int)(idx & (NB - 1));
    float d = dist[idx];
    float K = expf(-d * i2h2);
    s += K * gss[idx] + (gxsy[idx] - ysy[j]) * K * ih2 + d * K * ih4 - K * dimh;
  }
  __shared__ float sw[8];
#pragma unroll
  for (int m = 1; m < 32; m <<= 1) s += __shfl_xor(s, m, 32);
  if ((threadIdx.x & 31) == 0) sw[threadIdx.x >> 5] = s;
  __syncthreads();
  if (threadIdx.x == 0) {
    float tot = 0.f;
#pragma unroll
    for (int w = 0; w < 8; ++w) tot += sw[w];
    partial[blockIdx.x] = tot;
  }
}

// ---------------------------------------------------------------------------
// Final scalars: align, kl, total
// ---------------------------------------------------------------------------
__global__ __launch_bounds__(256) void finalize_kernel(
    const float* __restrict__ partial, const float* __restrict__ kl1,
    const float* __restrict__ kl2, float* __restrict__ out3) {
  int tid = threadIdx.x;
  float sa = 0.f, s1 = 0.f, s2 = 0.f;
  for (int i = tid; i < NB; i += 256) { sa += partial[i]; s1 += kl1[i]; s2 += kl2[i]; }
  __shared__ float wa[8], w1[8], w2[8];
#pragma unroll
  for (int m = 1; m < 32; m <<= 1) {
    sa += __shfl_xor(sa, m, 32);
    s1 += __shfl_xor(s1, m, 32);
    s2 += __shfl_xor(s2, m, 32);
  }
  if ((tid & 31) == 0) { wa[tid >> 5] = sa; w1[tid >> 5] = s1; w2[tid >> 5] = s2; }
  __syncthreads();
  if (tid == 0) {
    float ta = 0, t1 = 0, t2 = 0;
#pragma unroll
    for (int w = 0; w < 8; ++w) { ta += wa[w]; t1 += w1[w]; t2 += w2[w]; }
    float align = ta / ((float)NB * (float)NB);
    float kl = 0.5f * (t1 / (float)NB + t2 / (float)NB);
    out3[0] = align;
    out3[1] = kl;
    out3[2] = kl + 0.1f * align;
  }
}

// ---------------------------------------------------------------------------
extern "C" void kernel_launch(void* const* d_in, const int* in_sizes, int n_in,
                              void* d_out, int out_size, void* d_ws, size_t ws_size,
                              hipStream_t stream) {
  const float* img_feat = (const float*)d_in[0];
  const float* txt_feat = (const float*)d_in[1];
  const int*   mask     = (const int*)d_in[2];
  const float* eps_img  = (const float*)d_in[3];
  const float* eps_txt  = (const float*)d_in[4];
  const float* img_W1 = (const float*)d_in[5];
  const float* img_b1 = (const float*)d_in[6];
  const float* img_g  = (const float*)d_in[7];
  const float* img_be = (const float*)d_in[8];
  const float* img_W2 = (const float*)d_in[9];
  const float* img_b2 = (const float*)d_in[10];
  const float* txt_W1 = (const float*)d_in[11];
  const float* txt_b1 = (const float*)d_in[12];
  const float* txt_g  = (const float*)d_in[13];
  const float* txt_be = (const float*)d_in[14];
  const float* txt_W2 = (const float*)d_in[15];
  const float* txt_b2 = (const float*)d_in[16];
  const float* is_W1 = (const float*)d_in[17];
  const float* is_b1 = (const float*)d_in[18];
  const float* is_g  = (const float*)d_in[19];
  const float* is_be = (const float*)d_in[20];
  const float* is_W2 = (const float*)d_in[21];
  const float* is_b2 = (const float*)d_in[22];
  const float* ts_W1 = (const float*)d_in[23];
  const float* ts_b1 = (const float*)d_in[24];
  const float* ts_g  = (const float*)d_in[25];
  const float* ts_be = (const float*)d_in[26];
  const float* ts_W2 = (const float*)d_in[27];
  const float* ts_b2 = (const float*)d_in[28];

  float* out = (float*)d_out;
  float* img_z = out;                              // (B,S,256)
  float* txt_z = out + (size_t)NB * NS * ND;       // (B,S,256)
  float* out3  = out + 2 * (size_t)NB * NS * ND;   // align, kl, total

  // workspace layout
  char* ws = (char*)d_ws;
  size_t o = 0;
  auto alloc = [&](size_t bytes) { size_t r = o; o = (o + bytes + 255) & ~(size_t)255; return r; };
  half_t* w1t_img = (half_t*)(ws + alloc((size_t)1024 * 768 * 2));
  half_t* w2t_img = (half_t*)(ws + alloc((size_t)512 * 1024 * 2));
  half_t* w1t_txt = (half_t*)(ws + alloc((size_t)1024 * 768 * 2));
  half_t* w2t_txt = (half_t*)(ws + alloc((size_t)512 * 1024 * 2));
  half_t* isw1t   = (half_t*)(ws + alloc((size_t)512 * 256 * 2));
  half_t* isw2t   = (half_t*)(ws + alloc((size_t)256 * 512 * 2));
  half_t* tsw1t   = (half_t*)(ws + alloc((size_t)512 * 256 * 2));
  half_t* tsw2t   = (half_t*)(ws + alloc((size_t)256 * 512 * 2));
  int*    seq_len = (int*)(ws + alloc((size_t)NB * 4));
  float*  zsel_i  = (float*)(ws + alloc((size_t)NB * ND * 4));
  float*  zsel_t  = (float*)(ws + alloc((size_t)NB * ND * 4));
  float*  s_img   = (float*)(ws + alloc((size_t)NB * ND * 4));
  float*  s_txt   = (float*)(ws + alloc((size_t)NB * ND * 4));
  float*  dist    = (float*)(ws + alloc((size_t)NB * NB * 4));
  float*  gss     = (float*)(ws + alloc((size_t)NB * NB * 4));
  float*  gxsy    = (float*)(ws + alloc((size_t)NB * NB * 4));
  float*  xx      = (float*)(ws + alloc((size_t)NB * 4));
  float*  yy      = (float*)(ws + alloc((size_t)NB * 4));
  float*  ysy     = (float*)(ws + alloc((size_t)NB * 4));
  float*  h2      = (float*)(ws + alloc(256));
  float*  kl1     = (float*)(ws + alloc((size_t)NB * 4));
  float*  kl2     = (float*)(ws + alloc((size_t)NB * 4));
  float*  alignp  = (float*)(ws + alloc((size_t)NB * 4));

  // 1) weight prep
  auto tr = [&](const float* src, half_t* dst, int K, int N) {
    transpose_f16_kernel<<<(K * N + 255) / 256, 256, 0, stream>>>(src, dst, K, N);
  };
  tr(img_W1, w1t_img, 768, 1024);
  tr(img_W2, w2t_img, 1024, 512);
  tr(txt_W1, w1t_txt, 768, 1024);
  tr(txt_W2, w2t_txt, 1024, 512);
  tr(is_W1, isw1t, 256, 512);
  tr(is_W2, isw2t, 512, 256);
  tr(ts_W1, tsw1t, 256, 512);
  tr(ts_W2, tsw2t, 512, 256);

  // 2) seq_len
  seqlen_kernel<<<(NB + 255) / 256, 256, 0, stream>>>(mask, seq_len);

  // 3) main encoder MLPs (65536 tokens, 32 per block); dyn LDS = 32*HID*2 = 64KB
  const int ntok = NB * NS;
  const size_t lds_main  = (size_t)32 * 1024 * sizeof(half_t);
  const size_t lds_score = (size_t)32 * 512 * sizeof(half_t);
  fused_mlp_kernel<768, 1024, 512, true><<<ntok / 32, 256, lds_main, stream>>>(
      img_feat, w1t_img, img_b1, img_g, img_be, w2t_img, img_b2,
      eps_img, seq_len, img_z, zsel_i, kl1, nullptr);
  fused_mlp_kernel<768, 1024, 512, true><<<ntok / 32, 256, lds_main, stream>>>(
      txt_feat, w1t_txt, txt_b1, txt_g, txt_be, w2t_txt, txt_b2,
      eps_txt, seq_len, txt_z, zsel_t, kl2, nullptr);

  // 4) score MLPs on selected tokens only (1024 rows)
  fused_mlp_kernel<256, 512, 256, false><<<NB / 32, 256, lds_score, stream>>>(
      zsel_i, isw1t, is_b1, is_g, is_be, isw2t, is_b2,
      nullptr, nullptr, nullptr, nullptr, nullptr, s_img);
  fused_mlp_kernel<256, 512, 256, false><<<NB / 32, 256, lds_score, stream>>>(
      zsel_t, tsw1t, ts_b1, ts_g, ts_be, tsw2t, ts_b2,
      nullptr, nullptr, nullptr, nullptr, nullptr, s_txt);

  // 5) pairwise terms
  rownorm_kernel<<<NB, 256, 0, stream>>>(zsel_i, zsel_t, s_txt, xx, yy, ysy);
  pairwise_kernel<<<(64 * 64) / 8, 256, 0, stream>>>(
      zsel_i, s_img, zsel_t, s_txt, xx, yy, dist, gss, gxsy);

  // 6) median bandwidth, align terms, scalars
  bandwidth_kernel<<<1, 1024, 0, stream>>>(dist, h2);
  align_kernel<<<NB, 256, 0, stream>>>(dist, gss, gxsy, ysy, h2, alignp);
  finalize_kernel<<<1, 256, 0, stream>>>(alignp, kl1, kl2, out3);
}
